// FlashAttention_62551903699100
// MI455X (gfx1250) — matrix operand order repaired and checked by reading, not compiled, not run on hardware
//
#include <hip/hip_runtime.h>
#include <hip/hip_bf16.h>
#include <stdint.h>

// ---- CDNA5 WMMA types -------------------------------------------------------
typedef __attribute__((ext_vector_type(16))) __bf16 v16bf;
typedef __attribute__((ext_vector_type(8)))  float  v8f;
typedef __attribute__((ext_vector_type(4)))  int    v4i;

#define BATCH 4
#define HEADS 16
#define SEQ   4096
#define DH    128     // head dim
#define BK    128     // k-block (matches reference BLOCK_K)
#define MT    128     // q rows per workgroup (8 waves x 16 rows)

#if defined(__has_builtin)
#if __has_builtin(__builtin_amdgcn_cvt_pk_bf16_f32)
#define HAVE_PK_BF16 1
#endif
#if __has_builtin(__builtin_amdgcn_global_load_async_to_lds_b128)
#define HAVE_ASYNC_LDS 1
#endif
#endif
#ifndef HAVE_PK_BF16
#define HAVE_PK_BF16 0
#endif
#ifndef HAVE_ASYNC_LDS
#define HAVE_ASYNC_LDS 0
#endif

typedef __attribute__((address_space(1))) v4i* g_v4i_ptr;   // global int4*
typedef __attribute__((address_space(3))) v4i* l_v4i_ptr;   // LDS int4*

union Frag {
  v16bf v;
  unsigned short s[16];
  unsigned w[8];
  uint4 u[2];
};

// fp32 -> bf16 bits, round-to-nearest-even (scalar fallback)
__device__ __forceinline__ unsigned short f2bf(float f) {
  unsigned u = __float_as_uint(f);
  u += 0x7FFFu + ((u >> 16) & 1u);
  return (unsigned short)(u >> 16);
}

// two fp32 -> packed bf16x2 in one uint (v_cvt_pk_bf16_f32 when available)
__device__ __forceinline__ unsigned pk2bf(float x, float y) {
#if HAVE_PK_BF16
  auto r = __builtin_amdgcn_cvt_pk_bf16_f32(x, y);
  unsigned u;
  __builtin_memcpy(&u, &r, 4);
  return u;
#else
  return (unsigned)f2bf(x) | ((unsigned)f2bf(y) << 16);
#endif
}

// ---- pre-pass 1: K fp32 -> bf16, same row-major layout ----------------------
__global__ __launch_bounds__(256)
void conv_k_bf16(const float* __restrict__ K, unsigned short* __restrict__ Kb) {
  const size_t flat = ((size_t)blockIdx.x * 256 + threadIdx.x) * 8;
  const float4 a = *(const float4*)(K + flat);
  const float4 b = *(const float4*)(K + flat + 4);
  uint4 o;
  o.x = pk2bf(a.x, a.y);
  o.y = pk2bf(a.z, a.w);
  o.z = pk2bf(b.x, b.y);
  o.w = pk2bf(b.z, b.w);
  *(uint4*)(Kb + flat) = o;
}

// ---- pre-pass 2: V fp32 -> bf16, transposed per 128-block: [bh][blk][d][s] --
__global__ __launch_bounds__(256)
void conv_v_bf16_t(const float* __restrict__ V, unsigned short* __restrict__ Vt) {
  __shared__ unsigned short t_lds[BK * DH];
  const int tid = threadIdx.x;
  const size_t src = (size_t)blockIdx.x * (BK * DH);   // blockIdx = bh*32 + blk
  #pragma unroll
  for (int i = 0; i < 16; ++i) {
    const int idx = (i * 256 + tid) * 4;               // row-major [s][d]
    const float4 f = *(const float4*)(V + src + idx);
    uint2 p;
    p.x = pk2bf(f.x, f.y);
    p.y = pk2bf(f.z, f.w);
    *(uint2*)&t_lds[idx] = p;
  }
  __syncthreads();
  const int d  = tid >> 1;
  const int s0 = (tid & 1) * 64;
  unsigned short* dst = Vt + src + (size_t)d * BK + s0;
  #pragma unroll
  for (int j = 0; j < 8; ++j) {
    unsigned v[8];
    #pragma unroll
    for (int m = 0; m < 8; ++m) v[m] = t_lds[(s0 + j * 8 + m) * DH + d];
    uint4 o;
    o.x = v[0] | (v[1] << 16);
    o.y = v[2] | (v[3] << 16);
    o.z = v[4] | (v[5] << 16);
    o.w = v[6] | (v[7] << 16);
    *(uint4*)(dst + j * 8) = o;
  }
}

// ---- flash attention forward ------------------------------------------------
template <bool PRE>
__global__ __launch_bounds__(256, 1)
void fa_fwd_bf16_wmma(const float* __restrict__ Q, const float* __restrict__ K,
                      const float* __restrict__ V,
                      const unsigned short* __restrict__ Kb,
                      const unsigned short* __restrict__ Vt,
                      __hip_bfloat16* __restrict__ O) {
  // K block row-major [seq][d]; after S is computed this region is reused as P
  __shared__ unsigned short kp_lds[BK * DH];
  // V block column-major [d][seq]  -> contiguous B-fragment loads for P@V
  __shared__ unsigned short v_lds[DH * BK];

  const int tid  = threadIdx.x;
  const int wave = tid >> 5;
  const int lane = tid & 31;
  const int l16  = lane & 15;
  const int lh   = lane >> 4;          // 0: lanes 0-15, 1: lanes 16-31

  const int bh = blockIdx.x >> 5;      // 64 (b,h) pairs
  const int qt = blockIdx.x & 31;      // 32 q tiles of 128 rows
  const size_t base = (size_t)bh * (SEQ * DH);
  const int qrow0 = qt * MT + wave * 16;

  // ---- Q -> bf16 A-fragments (16x32 each), held in registers for all blocks
  Frag qa[4];
  #pragma unroll
  for (int kb = 0; kb < 4; ++kb) {
    const float* qp = Q + base + (size_t)(qrow0 + l16) * DH + kb * 32 + lh * 8;
    #pragma unroll
    for (int e = 0; e < 4; ++e) qa[kb].w[e]     = pk2bf(qp[2 * e],      qp[2 * e + 1]);
    #pragma unroll
    for (int e = 0; e < 4; ++e) qa[kb].w[4 + e] = pk2bf(qp[16 + 2 * e], qp[16 + 2 * e + 1]);
  }

  v8f oacc[8] = {};
  // softmax state kept in log2 domain: m2 = m * log2(e)
  float m2row[8], lrow[8];
  #pragma unroll
  for (int r = 0; r < 8; ++r) { m2row[r] = -14426.950408889634f; lrow[r] = -10000.0f; }

  const float sc2 = 0.12753102543552743f;  // log2(e)/sqrt(128)

  for (int blk = 0; blk < SEQ / BK; ++blk) {
    const int krow0 = blk * BK;

    if constexpr (PRE) {
      // ---- staging = pure bf16 copies (async global->LDS when available) ----
      const unsigned short* kb_src = Kb + base + (size_t)krow0 * DH;
      const unsigned short* vt_src = Vt + ((size_t)(bh * (SEQ / BK) + blk)) * (DH * BK);
      #pragma unroll
      for (int i = 0; i < 8; ++i) {
        const int o = (i * 256 + tid) * 8;   // 8 shorts = 16 bytes per copy
#if HAVE_ASYNC_LDS
        __builtin_amdgcn_global_load_async_to_lds_b128(
            (g_v4i_ptr)(kb_src + o), (l_v4i_ptr)&kp_lds[o], 0, 0);
        __builtin_amdgcn_global_load_async_to_lds_b128(
            (g_v4i_ptr)(vt_src + o), (l_v4i_ptr)&v_lds[o], 0, 0);
#else
        *(uint4*)&kp_lds[o] = *(const uint4*)(kb_src + o);
        *(uint4*)&v_lds[o]  = *(const uint4*)(vt_src + o);
#endif
      }
#if HAVE_ASYNC_LDS
      asm volatile("s_wait_asynccnt 0" ::: "memory");
#endif
      if (blk + 1 < SEQ / BK) {
        __builtin_prefetch((const char*)(kb_src + BK * DH) + tid * 128, 0, 1);
        __builtin_prefetch((const char*)(vt_src + DH * BK) + tid * 128, 0, 1);
      }
    } else {
      // ---- fallback: convert fp32->bf16 while staging ----
      #pragma unroll
      for (int i = 0; i < 16; ++i) {
        const int idx = (i * 256 + tid) * 4;       // 0..16383
        const int r = idx >> 7;                    // seq within block
        const int c = idx & 127;                   // d
        const float4 kf = *(const float4*)(K + base + (size_t)(krow0 + r) * DH + c);
        uint2 kp;
        kp.x = pk2bf(kf.x, kf.y);
        kp.y = pk2bf(kf.z, kf.w);
        *(uint2*)&kp_lds[r * DH + c] = kp;
        const float4 vf = *(const float4*)(V + base + (size_t)(krow0 + r) * DH + c);
        const unsigned v01 = pk2bf(vf.x, vf.y);
        const unsigned v23 = pk2bf(vf.z, vf.w);
        v_lds[(c + 0) * BK + r] = (unsigned short)v01;
        v_lds[(c + 1) * BK + r] = (unsigned short)(v01 >> 16);
        v_lds[(c + 2) * BK + r] = (unsigned short)v23;
        v_lds[(c + 3) * BK + r] = (unsigned short)(v23 >> 16);
      }
      if (blk + 1 < SEQ / BK) {
        const char* nk = (const char*)(K + base + (size_t)(krow0 + BK) * DH);
        const char* nv = (const char*)(V + base + (size_t)(krow0 + BK) * DH);
        __builtin_prefetch(nk + tid * 256, 0, 1);
        __builtin_prefetch(nv + tid * 256, 0, 1);
      }
    }
    __syncthreads();

    // ---- S = Q * K^T  (16x128 per wave, 32 WMMAs)
    v8f sacc[8] = {};
    #pragma unroll
    for (int kb = 0; kb < 4; ++kb) {
      #pragma unroll
      for (int nt = 0; nt < 8; ++nt) {
        Frag bf;  // B element (kd, n) = K[n][kd]; kd = kb*32 + lh*8 + e (+16 for e>=8)
        const unsigned short* kp = &kp_lds[(nt * 16 + l16) * DH + kb * 32 + lh * 8];
        bf.u[0] = *(const uint4*)(kp);
        bf.u[1] = *(const uint4*)(kp + 16);
        sacc[nt] = __builtin_amdgcn_wmma_f32_16x16x32_bf16(
            false, qa[kb].v, false, bf.v, (short)0, sacc[nt], false, false);
      }
    }

    // ---- online softmax in log2 domain (C layout: VGPR r -> rows r, r+8)
    float corr[8];
    #pragma unroll
    for (int r = 0; r < 8; ++r) {
      float mx = sacc[0][r];
      #pragma unroll
      for (int nt = 1; nt < 8; ++nt) mx = fmaxf(mx, sacc[nt][r]);
      mx *= sc2;
      #pragma unroll
      for (int off = 1; off < 16; off <<= 1)
        mx = fmaxf(mx, __shfl_xor(mx, off, 32));   // stays within each 16-lane half
      const float m2new = fmaxf(m2row[r], mx);
      corr[r] = __builtin_amdgcn_exp2f(m2row[r] - m2new);
      m2row[r] = m2new;
    }

    __syncthreads();   // all waves done reading K -> safe to overwrite with P

    float rsum[8] = {};
    #pragma unroll
    for (int nt = 0; nt < 8; ++nt) {
      #pragma unroll
      for (int r = 0; r < 8; ++r) {
        const float p = __builtin_amdgcn_exp2f(sacc[nt][r] * sc2 - m2row[r]);
        rsum[r] += p;
        kp_lds[(wave * 16 + r + lh * 8) * DH + nt * 16 + l16] = f2bf(p);
      }
    }
    #pragma unroll
    for (int r = 0; r < 8; ++r) {
      #pragma unroll
      for (int off = 1; off < 16; off <<= 1)
        rsum[r] += __shfl_xor(rsum[r], off, 32);
      lrow[r] = lrow[r] * corr[r] + rsum[r];
      #pragma unroll
      for (int nt = 0; nt < 8; ++nt) oacc[nt][r] *= corr[r];
    }

    // wave-local LDS RAW: our P stores must land before our A-fragment reloads
    asm volatile("s_wait_dscnt 0" ::: "memory");

    // ---- O += P * V  (32 WMMAs)
    #pragma unroll
    for (int kb = 0; kb < 4; ++kb) {
      Frag pa;  // A element (m,k): m = l16, k = kb*32 + lh*8 + e (+16 for e>=8)
      const unsigned short* pp = &kp_lds[(wave * 16 + l16) * DH + kb * 32 + lh * 8];
      pa.u[0] = *(const uint4*)(pp);
      pa.u[1] = *(const uint4*)(pp + 16);
      #pragma unroll
      for (int nt = 0; nt < 8; ++nt) {
        Frag vb;  // B element (k, n) = V[k][nt*16+l16]; contiguous in col-major LDS
        const unsigned short* vp = &v_lds[(nt * 16 + l16) * BK + kb * 32 + lh * 8];
        vb.u[0] = *(const uint4*)(vp);
        vb.u[1] = *(const uint4*)(vp + 16);
        oacc[nt] = __builtin_amdgcn_wmma_f32_16x16x32_bf16(
            false, pa.v, false, vb.v, (short)0, oacc[nt], false, false);
      }
    }
    __syncthreads();   // done with this block's LDS before next overwrite
  }

  // ---- epilogue: O / L, store bf16
  #pragma unroll
  for (int nt = 0; nt < 8; ++nt) {
    #pragma unroll
    for (int r = 0; r < 8; ++r) {
      const int m = qrow0 + r + lh * 8;
      const int d = nt * 16 + l16;
      O[base + (size_t)m * DH + d] = __float2bfloat16(oacc[nt][r] / lrow[r]);
    }
  }
}

extern "C" void kernel_launch(void* const* d_in, const int* in_sizes, int n_in,
                              void* d_out, int out_size, void* d_ws, size_t ws_size,
                              hipStream_t stream) {
  const float* Q = (const float*)d_in[0];
  const float* K = (const float*)d_in[1];
  const float* V = (const float*)d_in[2];
  __hip_bfloat16* O = (__hip_bfloat16*)d_out;

  const size_t elems = (size_t)BATCH * HEADS * SEQ * DH;   // 33,554,432
  const size_t need  = elems * 2u * 2u;                    // Kb + Vt, bf16 each
  const int grid = BATCH * HEADS * (SEQ / MT);             // 2048 workgroups

  if (ws_size >= need) {
    unsigned short* Kb = (unsigned short*)d_ws;
    unsigned short* Vt = Kb + elems;
    conv_k_bf16<<<(int)(elems / (256 * 8)), 256, 0, stream>>>(K, Kb);
    conv_v_bf16_t<<<BATCH * HEADS * (SEQ / BK), 256, 0, stream>>>(V, Vt);
    fa_fwd_bf16_wmma<true><<<grid, 256, 0, stream>>>(Q, K, V, Kb, Vt, O);
  } else {
    fa_fwd_bf16_wmma<false><<<grid, 256, 0, stream>>>(Q, K, V, nullptr, nullptr, O);
  }
}
